// MMDLoss_44272522887284
// MI455X (gfx1250) — compile-verified, matrix operand order checked by reading, not run on hardware
//
#include <hip/hip_runtime.h>
#include <hip/hip_bf16.h>

typedef __attribute__((ext_vector_type(16))) __bf16          v16bf;
typedef __attribute__((ext_vector_type(8)))  float           v8f;
typedef __attribute__((ext_vector_type(8)))  unsigned short  us8;
typedef __attribute__((ext_vector_type(4)))  unsigned short  us4;

#define N_TOTAL 8192
#define BS      4096
#define DIMS    128
#define LDA     136      // LDS row stride in bf16 elems (272B -> bank-conflict-free)
#define TILE_M  128
#define TILE_N  64

// workspace layout (float offsets)
#define OFF_SQ   0       // 8192 row squared norms
#define OFF_COL  8192    // 128 column sums
#define OFF_ACC  8320    // scalar accumulator
#define OFF_C    8328    // exp2 scale coefficient (widest bandwidth)
#define INIT_LEN 160     // floats zeroed starting at OFF_COL

__device__ __forceinline__ unsigned short f2bf(float f) {
  unsigned int u = __float_as_uint(f);
  u += 0x7FFFu + ((u >> 16) & 1u);   // round-to-nearest-even
  return (unsigned short)(u >> 16);
}

// Load a 16x32 bf16 WMMA A/B fragment from an LDS tile (row-major, stride LDA).
// Per ISA layout: lanes 0-15 hold row M=lane, K {kb+0..7, kb+16..23};
// lanes 16-31 hold row M=lane-16, K {kb+8..15, kb+24..31}.
__device__ __forceinline__ v16bf frag_ld(const unsigned short* row0, int lane) {
  const int hl = lane & 15;
  const int hb = lane >> 4;
  const unsigned short* p = row0 + hl * LDA + hb * 8;
  union { struct { us8 a, b; } s; v16bf v; } u;
  u.s.a = *(const us8*)(p);
  u.s.b = *(const us8*)(p + 16);
  return u.v;
}

// 5-bandwidth gaussian kernel sum: bandwidths are a geometric ladder, so
// exp(-L2/bw_i) = exp(-L2/bw_{i+1})^2 -> 1 transcendental + 4 squarings.
__device__ __forceinline__ float tile_sum(v8f acc, const float* sqr, float sqc,
                                          float c4) {
  float s = 0.0f;
  #pragma unroll
  for (int v = 0; v < 8; ++v) {
    float L2 = fmaxf(sqr[v] + sqc - 2.0f * acc[v], 0.0f);
    float t4 = __builtin_exp2f(L2 * c4);   // widest bandwidth (i=4)
    float t3 = t4 * t4;
    float t2 = t3 * t3;
    float t1 = t2 * t2;
    float t0 = t1 * t1;
    s += t0 + t1 + t2 + t3 + t4;
  }
  return s;
}

// ---------------------------------------------------------------- init
__global__ __launch_bounds__(256) void mmd_init_kernel(float* __restrict__ ws) {
  int t = threadIdx.x;
  if (t < INIT_LEN) ws[OFF_COL + t] = 0.0f;
}

// ------------------------------------------------- row norms + col sums
// grid 64 x 256: each block handles 128 rows; lane owns 4 dims.
__global__ __launch_bounds__(256) void mmd_rowstats_kernel(
    const float* __restrict__ src, const float* __restrict__ tgt,
    float* __restrict__ ws) {
  __shared__ float scol[8 * DIMS];
  const int tid  = threadIdx.x;
  const int lane = tid & 31;
  const int wave = tid >> 5;
  const int rowBase = blockIdx.x * 128;
  const float* base = (rowBase < BS) ? (src + (size_t)rowBase * DIMS)
                                     : (tgt + (size_t)(rowBase - BS) * DIMS);
  float ca0 = 0.f, ca1 = 0.f, ca2 = 0.f, ca3 = 0.f;
  for (int k = 0; k < 16; ++k) {
    int r = wave * 16 + k;
    float4 v = ((const float4*)(base + (size_t)r * DIMS))[lane];
    ca0 += v.x; ca1 += v.y; ca2 += v.z; ca3 += v.w;
    float s = v.x * v.x + v.y * v.y + v.z * v.z + v.w * v.w;
    for (int m = 16; m > 0; m >>= 1) s += __shfl_xor(s, m, 32);
    if (lane == 0) ws[OFF_SQ + rowBase + r] = s;
  }
  scol[wave * DIMS + lane * 4 + 0] = ca0;
  scol[wave * DIMS + lane * 4 + 1] = ca1;
  scol[wave * DIMS + lane * 4 + 2] = ca2;
  scol[wave * DIMS + lane * 4 + 3] = ca3;
  __syncthreads();
  if (tid < DIMS) {
    float t = 0.f;
    for (int w = 0; w < 8; ++w) t += scol[w * DIMS + tid];
    atomicAdd(&ws[OFF_COL + tid], t);
  }
}

// ----------------------------------------------- bandwidth finalization
// sum(L2) = 2n*sum(||x||^2) - 2*sum_d(colsum_d^2); write widest exp2 scale.
__global__ __launch_bounds__(256) void mmd_bw_kernel(float* __restrict__ ws) {
  __shared__ float red[256];
  const int tid = threadIdx.x;
  float s = 0.f;
  for (int k = 0; k < 32; ++k) s += ws[OFF_SQ + tid + k * 256];
  red[tid] = s;
  __syncthreads();
  for (int st = 128; st > 0; st >>= 1) {
    if (tid < st) red[tid] += red[tid + st];
    __syncthreads();
  }
  float sumSq = red[0];
  __syncthreads();
  float c = (tid < DIMS) ? ws[OFF_COL + tid] : 0.f;
  red[tid] = c * c;
  __syncthreads();
  for (int st = 128; st > 0; st >>= 1) {
    if (tid < st) red[tid] += red[tid + st];
    __syncthreads();
  }
  if (tid == 0) {
    float nF = (float)N_TOTAL;
    float sumL2 = 2.0f * nF * sumSq - 2.0f * red[0];
    float bw = sumL2 / (nF * nF - nF) * 0.25f;      // / KERNEL_MUL^(5//2)
    // widest bandwidth term: bw_4 = bw * 2^4
    ws[OFF_C] = -1.4426950408889634f / (bw * 16.0f + 1e-10f); // -log2e/(bw4+eps)
  }
}

// -------------------------------------------------- fused GEMM+epilogue
// Symmetric: only upper-triangle 128x128 super-blocks; off-diagonal weight 2.
__global__ __launch_bounds__(256) void mmd_tile_kernel(
    const float* __restrict__ src, const float* __restrict__ tgt,
    float* __restrict__ ws) {
  __shared__ unsigned short sA[TILE_M * LDA];   // 34816 B
  __shared__ unsigned short sB[TILE_N * LDA];   // 17408 B

  const int I = blockIdx.y;        // 128-row band
  const int J = blockIdx.x >> 1;   // 128-col super-band (two 64-col tiles)
  if (I > J) return;               // symmetry: skip lower triangle (pre-barrier)

  const int tid  = threadIdx.x;
  const int lane = tid & 31;
  const int wave = tid >> 5;
  const int row0 = I * TILE_M;
  const int col0 = blockIdx.x * TILE_N;

  const float* aBase = (row0 < BS) ? (src + (size_t)row0 * DIMS)
                                   : (tgt + (size_t)(row0 - BS) * DIMS);
  const float* bBase = (col0 < BS) ? (src + (size_t)col0 * DIMS)
                                   : (tgt + (size_t)(col0 - BS) * DIMS);

  // stage A (128x128 f32 -> bf16), rows contiguous so linear float4 index works
  #pragma unroll
  for (int c = 0; c < 16; ++c) {
    int idx = c * 256 + tid;                 // 0..4095
    float4 v = ((const float4*)aBase)[idx];
    int r = idx >> 5, c4i = idx & 31;
    us4 h = { f2bf(v.x), f2bf(v.y), f2bf(v.z), f2bf(v.w) };
    *(us4*)&sA[r * LDA + c4i * 4] = h;
  }
  // stage B (64x128)
  #pragma unroll
  for (int c = 0; c < 8; ++c) {
    int idx = c * 256 + tid;                 // 0..2047
    float4 v = ((const float4*)bBase)[idx];
    int r = idx >> 5, c4i = idx & 31;
    us4 h = { f2bf(v.x), f2bf(v.y), f2bf(v.z), f2bf(v.w) };
    *(us4*)&sB[r * LDA + c4i * 4] = h;
  }
  __syncthreads();

  const int wM = (wave & 3) * 32;
  const int wN = (wave >> 2) * 32;

  v8f acc00 = {0.f,0.f,0.f,0.f,0.f,0.f,0.f,0.f};
  v8f acc01 = acc00, acc10 = acc00, acc11 = acc00;

  #pragma unroll
  for (int kb = 0; kb < DIMS; kb += 32) {
    v16bf a0 = frag_ld(&sA[(wM +  0) * LDA + kb], lane);
    v16bf a1 = frag_ld(&sA[(wM + 16) * LDA + kb], lane);
    v16bf b0 = frag_ld(&sB[(wN +  0) * LDA + kb], lane);
    v16bf b1 = frag_ld(&sB[(wN + 16) * LDA + kb], lane);
    acc00 = __builtin_amdgcn_wmma_f32_16x16x32_bf16(false, a0, false, b0, (short)0, acc00, false, false);
    acc01 = __builtin_amdgcn_wmma_f32_16x16x32_bf16(false, a0, false, b1, (short)0, acc01, false, false);
    acc10 = __builtin_amdgcn_wmma_f32_16x16x32_bf16(false, a1, false, b0, (short)0, acc10, false, false);
    acc11 = __builtin_amdgcn_wmma_f32_16x16x32_bf16(false, a1, false, b1, (short)0, acc11, false, false);
  }

  // epilogue: L2 = sq_i + sq_j - 2*G, gaussian ladder via repeated squaring
  const float c4 = ws[OFF_C];
  const float* sq = ws + OFF_SQ;
  const int hl = lane & 15;
  const int hb = lane >> 4;

  float sqr[2][8];
  #pragma unroll
  for (int m = 0; m < 2; ++m)
    #pragma unroll
    for (int v = 0; v < 8; ++v)
      sqr[m][v] = sq[row0 + wM + m * 16 + hb * 8 + v];
  float sqc0 = sq[col0 + wN + hl];
  float sqc1 = sq[col0 + wN + 16 + hl];

  float lsum = 0.f;
  lsum += tile_sum(acc00, sqr[0], sqc0, c4);
  lsum += tile_sum(acc01, sqr[0], sqc1, c4);
  lsum += tile_sum(acc10, sqr[1], sqc0, c4);
  lsum += tile_sum(acc11, sqr[1], sqc1, c4);

  // quadrant sign (+ for XX/YY, - for XY/YX) x symmetry weight (2 off-diag)
  float scale = ((row0 < BS) == (col0 < BS)) ? 1.0f : -1.0f;
  if (I < J) scale *= 2.0f;

  __syncthreads();                 // done with sA fragment reads; reuse as scratch
  float* red = (float*)sA;
  red[tid] = lsum * scale;
  __syncthreads();
  for (int st = 128; st > 0; st >>= 1) {
    if (tid < st) red[tid] += red[tid + st];
    __syncthreads();
  }
  if (tid == 0) atomicAdd(&ws[OFF_ACC], red[0]);
}

// ---------------------------------------------------------------- output
__global__ void mmd_out_kernel(const float* __restrict__ ws,
                               float* __restrict__ out) {
  out[0] = ws[OFF_ACC] * (1.0f / ((float)BS * (float)BS));
}

extern "C" void kernel_launch(void* const* d_in, const int* in_sizes, int n_in,
                              void* d_out, int out_size, void* d_ws, size_t ws_size,
                              hipStream_t stream) {
  (void)in_sizes; (void)n_in; (void)out_size; (void)ws_size;
  const float* src = (const float*)d_in[0];
  const float* tgt = (const float*)d_in[1];
  float* ws  = (float*)d_ws;
  float* out = (float*)d_out;

  mmd_init_kernel<<<1, 256, 0, stream>>>(ws);
  mmd_rowstats_kernel<<<64, 256, 0, stream>>>(src, tgt, ws);
  mmd_bw_kernel<<<1, 256, 0, stream>>>(ws);
  dim3 grid(N_TOTAL / TILE_N, N_TOTAL / TILE_M);   // (128, 64); lower tri exits
  mmd_tile_kernel<<<grid, 256, 0, stream>>>(src, tgt, ws);
  mmd_out_kernel<<<1, 1, 0, stream>>>(ws, out);
}
